// ZModel_19920058318867
// MI455X (gfx1250) — matrix operand order repaired and checked by reading, not compiled, not run on hardware
//
#include <hip/hip_runtime.h>
#include <hip/hip_bf16.h>
#include <math.h>

// ---------------- model dims (compile-time) ----------------
#define LAYERS 2
#define HID    2048
#define NHEAD  16
#define DHEAD  128
#define FFD    8192
#define VOCAB  32000
#define SEQ    1024
#define BATCH  2
#define ROWS   (BATCH * SEQ)   // 2048 token rows

typedef __attribute__((ext_vector_type(16))) _Float16 h16;
typedef __attribute__((ext_vector_type(8)))  _Float16 h8;
typedef __attribute__((ext_vector_type(8)))  float    f32x8;

// =====================================================================
// WMMA GEMM: C[M,N] = A[M,K] @ B(+transB) (+ Cadd), fp32 in/out,
// f16 WMMA with f32 accumulation. M%128==0, N%128==0, K%32==0.
// 256 threads = 8 waves; block tile 128x128; wave tile 32x64; K-step 32.
// =====================================================================
__global__ __launch_bounds__(256) void k_gemm_wmma(
    const float* __restrict__ A, const float* __restrict__ B,
    const float* __restrict__ Cadd, float* __restrict__ C,
    int M, int N, int K, int transB)
{
  // 48-half row stride: rows start 16B-aligned (96B) for b128 LDS reads.
  __shared__ _Float16 As[128][48];   // As[m][k]
  __shared__ _Float16 Bs[128][48];   // Bs[n][k]  (B held K-contiguous per column)

  const int tid    = threadIdx.x;
  const int lane   = tid & 31;
  const int wv     = tid >> 5;            // 0..7
  const int waveM  = (wv >> 1) * 32;      // 0,32,64,96
  const int waveN  = (wv & 1) * 64;       // 0,64
  const int blockM = blockIdx.y * 128;
  const int blockN = blockIdx.x * 128;
  const int hsel   = lane >> 4;           // 0/1 : lane half-group
  const int lr     = lane & 15;

  f32x8 acc[2][4];
#pragma unroll
  for (int i = 0; i < 2; ++i)
#pragma unroll
    for (int j = 0; j < 4; ++j)
#pragma unroll
      for (int e = 0; e < 8; ++e) acc[i][j][e] = 0.0f;

  // staging coordinates
  const int arow  = tid >> 1;             // 0..127
  const int acol  = (tid & 1) * 16;       // 0/16
  const int bkrow = tid >> 3;             // 0..31  (non-trans B)
  const int bnb   = (tid & 7) * 16;       // 0..112
  const int bnrow = tid >> 1;             // 0..127 (trans B)
  const int bkb   = (tid & 1) * 16;       // 0/16

  for (int k0 = 0; k0 < K; k0 += 32) {
    __syncthreads();

    // ---- stage A tile: fp32 -> f16 into As[m][k] ----
    {
      const float4* s4 = (const float4*)(A + (size_t)(blockM + arow) * K + k0 + acol);
      float4 f0 = s4[0], f1 = s4[1], f2 = s4[2], f3 = s4[3];
      if (k0 + 32 < K) __builtin_prefetch((const float*)s4 + 32, 0, 0);
      h8 p0, p1;
      p0[0]=(_Float16)f0.x; p0[1]=(_Float16)f0.y; p0[2]=(_Float16)f0.z; p0[3]=(_Float16)f0.w;
      p0[4]=(_Float16)f1.x; p0[5]=(_Float16)f1.y; p0[6]=(_Float16)f1.z; p0[7]=(_Float16)f1.w;
      p1[0]=(_Float16)f2.x; p1[1]=(_Float16)f2.y; p1[2]=(_Float16)f2.z; p1[3]=(_Float16)f2.w;
      p1[4]=(_Float16)f3.x; p1[5]=(_Float16)f3.y; p1[6]=(_Float16)f3.z; p1[7]=(_Float16)f3.w;
      *(h8*)&As[arow][acol]     = p0;
      *(h8*)&As[arow][acol + 8] = p1;
    }

    // ---- stage B tile into Bs[n][k] ----
    if (!transB) {
      // B is [K,N] row-major: transpose while staging
      const float4* s4 = (const float4*)(B + (size_t)(k0 + bkrow) * N + blockN + bnb);
      float4 f0 = s4[0], f1 = s4[1], f2 = s4[2], f3 = s4[3];
      if (k0 + 32 < K)
        __builtin_prefetch(B + (size_t)(k0 + 32 + bkrow) * N + blockN + bnb, 0, 0);
      float t[16] = {f0.x,f0.y,f0.z,f0.w, f1.x,f1.y,f1.z,f1.w,
                     f2.x,f2.y,f2.z,f2.w, f3.x,f3.y,f3.z,f3.w};
#pragma unroll
      for (int i = 0; i < 16; ++i) Bs[bnb + i][bkrow] = (_Float16)t[i];
    } else {
      // B is [N,K] row-major (e.g. logits @ tok_emb^T): K-contiguous rows
      const float4* s4 = (const float4*)(B + (size_t)(blockN + bnrow) * K + k0 + bkb);
      float4 f0 = s4[0], f1 = s4[1], f2 = s4[2], f3 = s4[3];
      if (k0 + 32 < K) __builtin_prefetch((const float*)s4 + 32, 0, 0);
      h8 p0, p1;
      p0[0]=(_Float16)f0.x; p0[1]=(_Float16)f0.y; p0[2]=(_Float16)f0.z; p0[3]=(_Float16)f0.w;
      p0[4]=(_Float16)f1.x; p0[5]=(_Float16)f1.y; p0[6]=(_Float16)f1.z; p0[7]=(_Float16)f1.w;
      p1[0]=(_Float16)f2.x; p1[1]=(_Float16)f2.y; p1[2]=(_Float16)f2.z; p1[3]=(_Float16)f2.w;
      p1[4]=(_Float16)f3.x; p1[5]=(_Float16)f3.y; p1[6]=(_Float16)f3.z; p1[7]=(_Float16)f3.w;
      *(h8*)&Bs[bnrow][bkb]     = p0;
      *(h8*)&Bs[bnrow][bkb + 8] = p1;
    }
    __syncthreads();

    // ---- fragment loads (ISA 7.12.2 16-bit layouts) + 8 WMMAs ----
    h16 af[2], bf[4];
#pragma unroll
    for (int i = 0; i < 2; ++i) {
      // A: row m = base+lr; halves 0..7 => K = 8*hsel+0..7, halves 8..15 => K = 16+8*hsel+0..7
      const _Float16* ap = &As[waveM + i * 16 + lr][0];
      h8 lo = *(const h8*)(ap + 8 * hsel);
      h8 hi = *(const h8*)(ap + 16 + 8 * hsel);
      af[i] = __builtin_shufflevector(lo, hi, 0,1,2,3,4,5,6,7,8,9,10,11,12,13,14,15);
    }
#pragma unroll
    for (int j = 0; j < 4; ++j) {
      // B: column n = base+lr; halves 0..7 => K = 8*hsel+0..7, halves 8..15 => K = 16+8*hsel+0..7
      const _Float16* bp = &Bs[waveN + j * 16 + lr][0];
      h8 lo = *(const h8*)(bp + 8 * hsel);
      h8 hi = *(const h8*)(bp + 16 + 8 * hsel);
      bf[j] = __builtin_shufflevector(lo, hi, 0,1,2,3,4,5,6,7,8,9,10,11,12,13,14,15);
    }
#pragma unroll
    for (int i = 0; i < 2; ++i)
#pragma unroll
      for (int j = 0; j < 4; ++j)
        acc[i][j] = __builtin_amdgcn_wmma_f32_16x16x32_f16(
            false, af[i], false, bf[j], (short)0, acc[i][j], false, false);
  }

  // ---- epilogue: C layout VGPR r -> row = r + 8*(lane/16), col = lane%16 ----
#pragma unroll
  for (int i = 0; i < 2; ++i)
#pragma unroll
    for (int j = 0; j < 4; ++j) {
      const int col  = blockN + waveN + j * 16 + lr;
      const int rowb = blockM + waveM + i * 16 + 8 * hsel;
#pragma unroll
      for (int r = 0; r < 8; ++r) {
        size_t idx = (size_t)(rowb + r) * N + col;
        float vvv = acc[i][j][r];
        if (Cadd) vvv += Cadd[idx];
        C[idx] = vvv;
      }
    }
}

// =====================================================================
// Embedding gather: x[b,0,:] = z_emb[seg[b]]; x[b,s,:] = tok_emb[ids[b,s-1]]
// =====================================================================
__global__ __launch_bounds__(256) void k_embed(
    const int* __restrict__ seg, const int* __restrict__ ids,
    const float* __restrict__ z_emb, const float* __restrict__ tok_emb,
    float* __restrict__ x)
{
  size_t idx = (size_t)blockIdx.x * 256 + threadIdx.x;
  if (idx >= (size_t)ROWS * HID) return;
  int    hh = (int)(idx % HID);
  size_t r  = idx / HID;
  int    s  = (int)(r % SEQ);
  int    b  = (int)(r / SEQ);
  float val;
  if (s == 0) val = z_emb[(size_t)seg[b] * HID + hh];
  else        val = tok_emb[(size_t)ids[b * SEQ + s - 1] * HID + hh];
  x[idx] = val;
}

// =====================================================================
// RMSNorm: one block per row, wave32 shfl reduction
// =====================================================================
__global__ __launch_bounds__(256) void k_rmsnorm(
    const float* __restrict__ x, const float* __restrict__ w,
    float* __restrict__ out)
{
  const int row = blockIdx.x;
  const float* xr = x + (size_t)row * HID;
  float* orow = out + (size_t)row * HID;
  __shared__ float red[8];
  __shared__ float sscale;
  float ss = 0.0f;
  for (int i = threadIdx.x; i < HID; i += 256) { float vv = xr[i]; ss += vv * vv; }
  for (int off = 16; off; off >>= 1) ss += __shfl_xor(ss, off, 32);
  if ((threadIdx.x & 31) == 0) red[threadIdx.x >> 5] = ss;
  __syncthreads();
  if (threadIdx.x == 0) {
    float t = 0.0f;
    for (int i = 0; i < 8; ++i) t += red[i];
    sscale = rsqrtf(t / (float)HID + 1e-6f);
  }
  __syncthreads();
  float sc = sscale;
  for (int i = threadIdx.x; i < HID; i += 256) orow[i] = xr[i] * sc * w[i];
}

// =====================================================================
// RoPE (in-place on q and k): half = 64, theta = 10000
// =====================================================================
__global__ __launch_bounds__(256) void k_rope(float* __restrict__ q, float* __restrict__ k)
{
  size_t idx = (size_t)blockIdx.x * 256 + threadIdx.x;
  if (idx >= (size_t)ROWS * NHEAD * (DHEAD / 2)) return;
  int    d    = (int)(idx & 63);
  size_t t    = idx >> 6;
  int    head = (int)(t & (NHEAD - 1));
  size_t row  = t >> 4;                       // b*SEQ + s
  int    s    = (int)(row & (SEQ - 1));
  float inv = __expf(-9.2103403719761836f * ((float)d / 64.0f));  // 10000^{-d/64}
  float sn, cs;
  __sincosf((float)s * inv, &sn, &cs);
  size_t base = row * HID + (size_t)head * DHEAD;
  float q1 = q[base + d], q2 = q[base + d + 64];
  q[base + d]      = q1 * cs - q2 * sn;
  q[base + d + 64] = q2 * cs + q1 * sn;
  float k1 = k[base + d], k2 = k[base + d + 64];
  k[base + d]      = k1 * cs - k2 * sn;
  k[base + d + 64] = k2 * cs + k1 * sn;
}

// =====================================================================
// Flash-style causal attention, fp32 (only ~2% of total FLOPs).
// Block = (32 queries) of one (b,h); K/V tiles of 16 in LDS (<64KB static).
// =====================================================================
#define QT 32
#define KT 16
__global__ __launch_bounds__(256) void k_attn(
    const float* __restrict__ q, const float* __restrict__ k,
    const float* __restrict__ v, const int* __restrict__ amask,
    float* __restrict__ o)
{
  __shared__ float Qs[QT][DHEAD];
  __shared__ float Ks[KT][DHEAD + 1];   // +1: avoid bank-broadcast conflicts on QK^T
  __shared__ float Vs[KT][DHEAD];
  __shared__ float Os[QT][DHEAD];
  __shared__ float Ps[QT][KT];
  __shared__ float mrow[QT], lrow[QT], crow[QT];

  const int tid = threadIdx.x;
  const int qt  = blockIdx.x;
  const int b   = blockIdx.y / NHEAD;
  const int h   = blockIdx.y % NHEAD;
  const size_t rowBase = (size_t)b * SEQ;
  const size_t hoff    = (size_t)h * DHEAD;

  for (int i = tid; i < QT * DHEAD; i += 256) {
    int r = i >> 7, d = i & 127;
    Qs[r][d] = q[(rowBase + qt * QT + r) * HID + hoff + d];
    Os[r][d] = 0.0f;
  }
  if (tid < QT) { mrow[tid] = -3.0e38f; lrow[tid] = 0.0f; }
  __syncthreads();

  const float scale = 0.08838834764831845f;   // 1/sqrt(128)
  const int kmax = qt * QT + QT;              // causal upper bound for this tile
  for (int kt0 = 0; kt0 < kmax; kt0 += KT) {
    for (int i = tid; i < KT * DHEAD; i += 256) {
      int r = i >> 7, d = i & 127;
      size_t off = (rowBase + kt0 + r) * HID + hoff + d;
      Ks[r][d] = k[off];
      Vs[r][d] = v[off];
    }
    __syncthreads();
    for (int i = tid; i < QT * KT; i += 256) {
      int qr = i / KT, kr = i % KT;
      int qpos = qt * QT + qr, kpos = kt0 + kr;
      float sc = -1.0e30f;
      if (kpos <= qpos) {
        int ok = (kpos == 0) ? 1 : amask[b * SEQ + kpos - 1];
        if (ok > 0) {
          float dot = 0.0f;
#pragma unroll 16
          for (int d = 0; d < DHEAD; ++d) dot += Qs[qr][d] * Ks[kr][d];
          sc = dot * scale;
        }
      }
      Ps[qr][kr] = sc;
    }
    __syncthreads();
    if (tid < QT) {   // wave 0: per-row online-softmax bookkeeping
      float mold = mrow[tid], mnew = mold;
#pragma unroll
      for (int j = 0; j < KT; ++j) mnew = fmaxf(mnew, Ps[tid][j]);
      float corr = __expf(mold - mnew);
      float lsum = lrow[tid] * corr;
#pragma unroll
      for (int j = 0; j < KT; ++j) {
        float p = __expf(Ps[tid][j] - mnew);
        Ps[tid][j] = p;
        lsum += p;
      }
      mrow[tid] = mnew; lrow[tid] = lsum; crow[tid] = corr;
    }
    __syncthreads();
    for (int i = tid; i < QT * DHEAD; i += 256) {
      int r = i >> 7, d = i & 127;
      float a = Os[r][d] * crow[r];
#pragma unroll
      for (int j = 0; j < KT; ++j) a += Ps[r][j] * Vs[j][d];
      Os[r][d] = a;
    }
    __syncthreads();
  }
  for (int i = tid; i < QT * DHEAD; i += 256) {
    int r = i >> 7, d = i & 127;
    o[(rowBase + qt * QT + r) * HID + hoff + d] = Os[r][d] / lrow[r];
  }
}

// =====================================================================
// SwiGLU elementwise: out = silu(g) * u  (out may alias g)
// =====================================================================
__global__ __launch_bounds__(256) void k_silu_mul(
    const float* __restrict__ g, const float* __restrict__ u,
    float* __restrict__ o, int n)
{
  int idx = blockIdx.x * 256 + threadIdx.x;
  if (idx < n) {
    float gv = g[idx];
    o[idx] = gv / (1.0f + __expf(-gv)) * u[idx];
  }
}

// =====================================================================
// Per-row NLL over V=32000 logits (no atomics: deterministic)
// =====================================================================
__global__ __launch_bounds__(256) void k_rowloss(
    const float* __restrict__ logits, const int* __restrict__ ids,
    const int* __restrict__ amask, float* __restrict__ nll, float* __restrict__ valid)
{
  const int row = blockIdx.x;
  const float* lr = logits + (size_t)row * VOCAB;
  __shared__ float red[8];
  __shared__ float sval;
  float mx = -3.0e38f;
  for (int i = threadIdx.x; i < VOCAB; i += 256) mx = fmaxf(mx, lr[i]);
  for (int off = 16; off; off >>= 1) mx = fmaxf(mx, __shfl_xor(mx, off, 32));
  if ((threadIdx.x & 31) == 0) red[threadIdx.x >> 5] = mx;
  __syncthreads();
  if (threadIdx.x == 0) {
    float t = red[0];
    for (int i = 1; i < 8; ++i) t = fmaxf(t, red[i]);
    sval = t;
  }
  __syncthreads();
  mx = sval;
  float se = 0.0f;
  for (int i = threadIdx.x; i < VOCAB; i += 256) se += __expf(lr[i] - mx);
  for (int off = 16; off; off >>= 1) se += __shfl_xor(se, off, 32);
  __syncthreads();
  if ((threadIdx.x & 31) == 0) red[threadIdx.x >> 5] = se;
  __syncthreads();
  if (threadIdx.x == 0) {
    float t = 0.0f;
    for (int i = 0; i < 8; ++i) t += red[i];
    nll[row]   = -(lr[ids[row]] - mx - logf(t));
    valid[row] = (amask[row] > 0) ? 1.0f : 0.0f;
  }
}

__global__ __launch_bounds__(256) void k_loss_final(
    const float* __restrict__ nll, const float* __restrict__ valid,
    int n, float* __restrict__ out)
{
  float s = 0.0f, c = 0.0f;
  for (int i = threadIdx.x; i < n; i += 256) { s += nll[i] * valid[i]; c += valid[i]; }
  for (int off = 16; off; off >>= 1) { s += __shfl_xor(s, off, 32); c += __shfl_xor(c, off, 32); }
  __shared__ float rs[8], rc[8];
  if ((threadIdx.x & 31) == 0) { rs[threadIdx.x >> 5] = s; rc[threadIdx.x >> 5] = c; }
  __syncthreads();
  if (threadIdx.x == 0) {
    float ts = 0.0f, tc = 0.0f;
    for (int i = 0; i < 8; ++i) { ts += rs[i]; tc += rc[i]; }
    out[0] = ts / fmaxf(tc, 1.0f);
  }
}

// =====================================================================
// Host orchestration (graph-capture safe: launches only)
// =====================================================================
extern "C" void kernel_launch(void* const* d_in, const int* in_sizes, int n_in,
                              void* d_out, int out_size, void* d_ws, size_t ws_size,
                              hipStream_t stream) {
  (void)in_sizes; (void)n_in; (void)out_size; (void)ws_size;
  const int*   seg_idx   = (const int*)d_in[0];
  const int*   input_ids = (const int*)d_in[1];
  const int*   amask     = (const int*)d_in[2];
  const float* z_emb     = (const float*)d_in[3];
  const float* tok_emb   = (const float*)d_in[4];
  const float* wq        = (const float*)d_in[5];
  const float* wk        = (const float*)d_in[6];
  const float* wv        = (const float*)d_in[7];
  const float* wo        = (const float*)d_in[8];
  const float* w_gate    = (const float*)d_in[9];
  const float* w_up      = (const float*)d_in[10];
  const float* w_down    = (const float*)d_in[11];
  const float* ln1       = (const float*)d_in[12];
  const float* ln2       = (const float*)d_in[13];
  const float* ln_f      = (const float*)d_in[14];

  float* outp   = (float*)d_out;   // [0] = loss
  float* logits = outp + 1;        // [ROWS * VOCAB]

  // workspace carve-up (~235 MB of fp32)
  float* ws = (float*)d_ws;
  const size_t MH = (size_t)ROWS * HID;
  const size_t MF = (size_t)ROWS * FFD;
  float* x    = ws; ws += MH;
  float* xn   = ws; ws += MH;
  float* qb   = ws; ws += MH;
  float* kb   = ws; ws += MH;
  float* vb   = ws; ws += MH;
  float* ob   = ws; ws += MH;
  float* gb   = ws; ws += MF;
  float* ub   = ws; ws += MF;
  float* nllb = ws; ws += ROWS;
  float* valb = ws; ws += ROWS;

  dim3 blk(256);
  auto gemm = [&](const float* Ap, const float* Bp, const float* Cadd, float* Cp,
                  int Mm, int Nn, int Kk, int tB) {
    dim3 grid(Nn / 128, Mm / 128);
    k_gemm_wmma<<<grid, blk, 0, stream>>>(Ap, Bp, Cadd, Cp, Mm, Nn, Kk, tB);
  };

  k_embed<<<(unsigned)((MH + 255) / 256), blk, 0, stream>>>(seg_idx, input_ids, z_emb, tok_emb, x);

  for (int l = 0; l < LAYERS; ++l) {
    const size_t wHH = (size_t)l * HID * HID;
    const size_t wHF = (size_t)l * HID * FFD;
    k_rmsnorm<<<ROWS, blk, 0, stream>>>(x, ln1 + (size_t)l * HID, xn);
    gemm(xn, wq + wHH, nullptr, qb, ROWS, HID, HID, 0);
    gemm(xn, wk + wHH, nullptr, kb, ROWS, HID, HID, 0);
    gemm(xn, wv + wHH, nullptr, vb, ROWS, HID, HID, 0);
    size_t nrope = (size_t)ROWS * NHEAD * (DHEAD / 2);
    k_rope<<<(unsigned)((nrope + 255) / 256), blk, 0, stream>>>(qb, kb);
    dim3 agrid(SEQ / QT, BATCH * NHEAD);
    k_attn<<<agrid, blk, 0, stream>>>(qb, kb, vb, amask, ob);
    gemm(ob, wo + wHH, x, x, ROWS, HID, HID, 0);                       // residual add
    k_rmsnorm<<<ROWS, blk, 0, stream>>>(x, ln2 + (size_t)l * HID, xn);
    gemm(xn, w_gate + wHF, nullptr, gb, ROWS, FFD, HID, 0);
    gemm(xn, w_up + wHF, nullptr, ub, ROWS, FFD, HID, 0);
    k_silu_mul<<<(unsigned)((MF + 255) / 256), blk, 0, stream>>>(gb, ub, gb, (int)MF);
    gemm(gb, w_down + wHF, x, x, ROWS, HID, FFD, 0);                   // residual add
  }

  k_rmsnorm<<<ROWS, blk, 0, stream>>>(x, ln_f, xn);
  gemm(xn, tok_emb, nullptr, logits, ROWS, VOCAB, HID, 1);             // B^T path
  k_rowloss<<<ROWS, blk, 0, stream>>>(logits, input_ids, amask, nllb, valb);
  k_loss_final<<<1, blk, 0, stream>>>(nllb, valb, ROWS, outp);
}
